// DecoderTransformer_32779190403393
// MI455X (gfx1250) — matrix operand order repaired and checked by reading, not compiled, not run on hardware
//
#include <hip/hip_runtime.h>
#include <hip/hip_bf16.h>
#include <math.h>

// ---------------------------------------------------------------------------
// GPT-2-style decoder forward for gfx1250 (MI455X), WMMA bf16 everywhere.
//   B=2, T=1024, C=1024, H=16, HD=64, L=6, VOCAB=50257
// Compute-bound (~570 GFLOP vs ~21us of HBM traffic at 23.3 TB/s), so all
// GEMMs go through v_wmma_f32_16x16x32_bf16 (fp32 accumulate).
// ---------------------------------------------------------------------------

typedef __bf16 bf16_t;
typedef __attribute__((ext_vector_type(16))) __bf16 v16bf;
typedef __attribute__((ext_vector_type(8)))  __bf16 v8bf;
typedef __attribute__((ext_vector_type(8)))  float  v8f;
typedef __attribute__((ext_vector_type(16))) float  v16f;

#define GPT_B 2
#define GPT_T 1024
#define GPT_C 1024
#define GPT_H 16
#define GPT_HD 64
#define GPT_L 6
#define GPT_V 50257
#define GPT_M (GPT_B * GPT_T)   // 2048 rows of activations

// ---------------- WMMA helpers (layouts per CDNA5 ISA 7.12.2) ----------------

__device__ inline v8f wmma_bf16(v16bf a, v16bf b, v8f c) {
  return __builtin_amdgcn_wmma_f32_16x16x32_bf16(false, a, false, b, (short)0, c,
                                                 false, false);
}

__device__ inline v16bf concat8(v8bf lo, v8bf hi) {
  return __builtin_shufflevector(lo, hi, 0, 1, 2, 3, 4, 5, 6, 7, 8, 9, 10, 11,
                                 12, 13, 14, 15);
}

// A matrix fragment, 16x32 bf16, row M = lane&15.
// elems j -> K = k0 + (lane>=16 ? 16 : 0) + j   (contiguous 16 along K)
__device__ inline v16bf frag_a_bf16(const bf16_t* base, int ld, int row, int k0,
                                    int lane) {
  int kb = k0 + (lane & 16);
  const bf16_t* p = base + (size_t)row * ld + kb;
  return concat8(*(const v8bf*)p, *(const v8bf*)(p + 8));
}

// B matrix fragment, 32x16, column N = lane&15.
// elems j -> K = k0 + (lane>=16 ? 16 : 0) + j   (contiguous 16 along K)
// Source row-major N x K (weight row n contiguous in K), fp32 -> bf16 packed.
__device__ inline v16bf frag_b_f32(const float* base, int ld, int nrow, int k0,
                                   int lane) {
  int kb = k0 + (lane & 16);
  const float* p = base + (size_t)nrow * ld + kb;
  v16f v = *(const v16f*)p;                 // 64B aligned by construction
  return __builtin_convertvector(v, v16bf); // v_cvt_pk_bf16_f32 pairs
}

// Same B fragment but bf16 source (16 contiguous bf16 = 32B, aligned).
__device__ inline v16bf frag_b_bf16(const bf16_t* base, int ld, int nrow, int k0,
                                    int lane) {
  int kb = k0 + (lane & 16);
  const bf16_t* p = base + (size_t)nrow * ld + kb;
  return *(const v16bf*)p;
}

// ---------------------------- embedding ------------------------------------

__global__ void embed_kernel(const int* __restrict__ idx,
                             const float* __restrict__ tok,
                             const float* __restrict__ pos,
                             float* __restrict__ x) {
  size_t i = (size_t)blockIdx.x * 256 + threadIdx.x;
  if (i >= (size_t)GPT_M * GPT_C) return;
  int c = (int)(i & (GPT_C - 1));
  size_t bt = i >> 10;              // C == 1024
  int t = (int)(bt & (GPT_T - 1));
  x[i] = tok[(size_t)idx[bt] * GPT_C + c] + pos[(size_t)t * GPT_C + c];
}

// ---------------------------- layernorm -> bf16 -----------------------------

__global__ void __launch_bounds__(128)
ln_kernel(const float* __restrict__ x, const float* __restrict__ w,
          const float* __restrict__ b, bf16_t* __restrict__ out) {
  const int C = GPT_C;
  const int tid = threadIdx.x;
  const size_t row = blockIdx.x;
  const float* xr = x + row * C;
  __shared__ float red[128];

  float s = 0.f;
  for (int c = tid; c < C; c += 128) s += xr[c];
  red[tid] = s;
  __syncthreads();
#pragma unroll
  for (int off = 64; off > 0; off >>= 1) {
    if (tid < off) red[tid] += red[tid + off];
    __syncthreads();
  }
  float mean = red[0] * (1.0f / C);
  __syncthreads();

  float v = 0.f;
  for (int c = tid; c < C; c += 128) {
    float d = xr[c] - mean;
    v += d * d;
  }
  red[tid] = v;
  __syncthreads();
#pragma unroll
  for (int off = 64; off > 0; off >>= 1) {
    if (tid < off) red[tid] += red[tid + off];
    __syncthreads();
  }
  float rstd = rsqrtf(red[0] * (1.0f / C) + 1e-5f);

  for (int c = tid; c < C; c += 128)
    out[row * C + c] = (bf16_t)((xr[c] - mean) * rstd * w[c] + b[c]);
}

// ------------------- generic WMMA GEMM: D = A(bf16) * W(f32)^T --------------
// A: M x K bf16 row-major (lda).  W: N x K fp32 row-major (ldw).
// Each wave: 16(M) x 64(N) strip; block = 8 waves -> 128(M) x 64(N).

#define GOP_BF16  0   // out bf16
#define GOP_GELU  1   // out bf16, exact GELU applied
#define GOP_RESID 2   // out f32 = resid + acc
#define GOP_F32   3   // out f32

template <int OP>
__global__ void __launch_bounds__(256)
gemm_wmma(const bf16_t* __restrict__ A, int lda, const float* __restrict__ W,
          int ldw, int M, int N, int K, void* __restrict__ OutV, int ldo,
          const float* __restrict__ resid) {
  const int lane = threadIdx.x & 31;
  const int wave = threadIdx.x >> 5;
  const int m0 = blockIdx.y * 128 + wave * 16;
  const int n0 = blockIdx.x * 64;
  if (m0 >= M) return;

  v8f acc[4] = {};
  const int arow = m0 + (lane & 15);

#pragma unroll 2
  for (int k0 = 0; k0 < K; k0 += 32) {
    v16bf af = frag_a_bf16(A, lda, arow, k0, lane);
#pragma unroll
    for (int t = 0; t < 4; ++t) {
      int wr = n0 + t * 16 + (lane & 15);
      if (wr >= N) wr = N - 1;  // clamp; garbage columns are never stored
      v16bf bf = frag_b_f32(W, ldw, wr, k0, lane);
      acc[t] = wmma_bf16(af, bf, acc[t]);
    }
  }

#pragma unroll
  for (int t = 0; t < 4; ++t) {
    int col = n0 + t * 16 + (lane & 15);
    if (col >= N) continue;
#pragma unroll
    for (int r = 0; r < 8; ++r) {
      int row = m0 + r + ((lane & 16) >> 1);
      size_t o = (size_t)row * ldo + col;
      float v = acc[t][r];
      if constexpr (OP == GOP_BF16) {
        ((bf16_t*)OutV)[o] = (bf16_t)v;
      } else if constexpr (OP == GOP_GELU) {
        float g = 0.5f * v * (1.0f + erff(v * 0.70710678118f));
        ((bf16_t*)OutV)[o] = (bf16_t)g;
      } else if constexpr (OP == GOP_RESID) {
        ((float*)OutV)[o] = resid[o] + v;
      } else {
        ((float*)OutV)[o] = v;
      }
    }
  }
}

// ------------------------- flash attention (per wave) -----------------------
// One wave per (b, h, 16-row q tile). Streams 32-wide key tiles with running
// softmax; O accumulated in 4 WMMA fragments (HD=64). qkv: (B,T,3C) bf16.

__global__ void __launch_bounds__(32)
attn_kernel(const bf16_t* __restrict__ qkv, bf16_t* __restrict__ y) {
  const int lane = threadIdx.x & 31;
  const int m = lane & 15;
  const int lh8 = (lane & 16) >> 1;   // 0 or 8
  const int qt = blockIdx.x & 63;                 // T/16
  const int h = (blockIdx.x >> 6) & 15;
  const int b = blockIdx.x >> 10;
  const int C3 = 3 * GPT_C;

  const bf16_t* Qb = qkv + (size_t)b * GPT_T * C3 + h * GPT_HD;
  const bf16_t* Kb = Qb + GPT_C;
  const bf16_t* Vb = Qb + 2 * GPT_C;

  const int q0 = qt * 16;
  v16bf qf0 = frag_a_bf16(Qb, C3, q0 + m, 0, lane);
  v16bf qf1 = frag_a_bf16(Qb, C3, q0 + m, 32, lane);

  v8f oacc[4] = {};
  __shared__ float S[16][32];
  __shared__ bf16_t P[16][32];
  __shared__ float mrow[16], lrow[16], arow[16];
  if (lane < 16) { mrow[lane] = -1e30f; lrow[lane] = 0.f; }
  __syncthreads();

  const int qend = q0 + 15;
  for (int s0 = 0; s0 <= qend; s0 += 32) {
    // --- S = scale * Q K^T for two 16-wide subtiles, causal-masked ----------
#pragma unroll
    for (int sub = 0; sub < 2; ++sub) {
      int sb = s0 + sub * 16;
      int krow = sb + m;
      if (krow > GPT_T - 1) krow = GPT_T - 1;    // masked anyway
      v16bf kf0 = frag_b_bf16(Kb, C3, krow, 0, lane);
      v16bf kf1 = frag_b_bf16(Kb, C3, krow, 32, lane);
      v8f sacc = {};
      sacc = wmma_bf16(qf0, kf0, sacc);
      sacc = wmma_bf16(qf1, kf1, sacc);
#pragma unroll
      for (int r = 0; r < 8; ++r) {
        int row = r + lh8;                 // query row within tile
        float val = sacc[r] * 0.125f;      // 1/sqrt(64)
        if (sb + m > q0 + row) val = -1e30f;
        S[row][sub * 16 + m] = val;
      }
    }
    __syncthreads();

    // --- running softmax (lanes 0..15 each own one query row) ---------------
    if (lane < 16) {
      int rw = lane;
      float mx = mrow[rw];
      float tmax = -1e30f;
#pragma unroll
      for (int c = 0; c < 32; ++c) tmax = fmaxf(tmax, S[rw][c]);
      float mnew = fmaxf(mx, tmax);
      float alpha = __expf(mx - mnew);
      float sum = 0.f;
#pragma unroll
      for (int c = 0; c < 32; ++c) {
        float e = __expf(S[rw][c] - mnew);
        P[rw][c] = (bf16_t)e;
        sum += e;
      }
      lrow[rw] = lrow[rw] * alpha + sum;
      mrow[rw] = mnew;
      arow[rw] = alpha;
    }
    __syncthreads();

    // --- O = O*alpha + P V  (K = 32 over key tokens) ------------------------
    v16bf pf;
    {
      const bf16_t* pp = &P[m][lane & 16];
      pf = concat8(*(const v8bf*)pp, *(const v8bf*)(pp + 8));
    }
#pragma unroll
    for (int ch = 0; ch < 4; ++ch) {
      v16bf vf;
#pragma unroll
      for (int j = 0; j < 16; ++j) {
        int k = (lane & 16) + j;
        vf[j] = Vb[(size_t)(s0 + k) * C3 + ch * 16 + m];
      }
#pragma unroll
      for (int r = 0; r < 8; ++r) oacc[ch][r] *= arow[r + lh8];
      oacc[ch] = wmma_bf16(pf, vf, oacc[ch]);
    }
  }

  // --- normalize and store -------------------------------------------------
#pragma unroll
  for (int ch = 0; ch < 4; ++ch) {
#pragma unroll
    for (int r = 0; r < 8; ++r) {
      int row = r + lh8;
      float inv = 1.0f / lrow[row];
      y[(size_t)(b * GPT_T + q0 + row) * GPT_C + h * GPT_HD + ch * 16 + m] =
          (bf16_t)(oacc[ch][r] * inv);
    }
  }
}

// ------------------------------- host side ----------------------------------

extern "C" void kernel_launch(void* const* d_in, const int* in_sizes, int n_in,
                              void* d_out, int out_size, void* d_ws,
                              size_t ws_size, hipStream_t stream) {
  const int* idx = (const int*)d_in[0];
  const float* tok_emb = (const float*)d_in[1];
  const float* pos_emb = (const float*)d_in[2];
  const float* ln1_w = (const float*)d_in[3];
  const float* ln1_b = (const float*)d_in[4];
  const float* attn_w = (const float*)d_in[5];
  const float* proj_w = (const float*)d_in[6];
  const float* ln2_w = (const float*)d_in[7];
  const float* ln2_b = (const float*)d_in[8];
  const float* fc1_w = (const float*)d_in[9];
  const float* fc2_w = (const float*)d_in[10];
  const float* lnf_w = (const float*)d_in[11];
  const float* lnf_b = (const float*)d_in[12];

  // workspace layout (all 256B aligned by construction) ~44 MB total
  char* ws = (char*)d_ws;
  float* x = (float*)ws;                                        // M*C f32
  bf16_t* hbuf = (bf16_t*)(ws + (size_t)GPT_M * GPT_C * 4);     // M*C bf16
  bf16_t* qkv = (bf16_t*)((char*)hbuf + (size_t)GPT_M * GPT_C * 2);   // M*3C
  bf16_t* ybuf = (bf16_t*)((char*)qkv + (size_t)GPT_M * 3 * GPT_C * 2);  // M*C
  bf16_t* a1 = (bf16_t*)((char*)ybuf + (size_t)GPT_M * GPT_C * 2);  // M*4C

  const dim3 blk256(256), blk128(128), blk32(32);
  const int mblocks = GPT_M / 128;  // 16

  // x = tok_emb[idx] + pos_emb
  {
    size_t total = (size_t)GPT_M * GPT_C;
    embed_kernel<<<dim3((total + 255) / 256), blk256, 0, stream>>>(idx, tok_emb,
                                                                   pos_emb, x);
  }

  for (int l = 0; l < GPT_L; ++l) {
    const float* aw = attn_w + (size_t)l * 3 * GPT_C * GPT_C;
    const float* pw = proj_w + (size_t)l * GPT_C * GPT_C;
    const float* f1 = fc1_w + (size_t)l * 4 * GPT_C * GPT_C;
    const float* f2 = fc2_w + (size_t)l * GPT_C * 4 * GPT_C;

    // h = LN1(x) (bf16)
    ln_kernel<<<dim3(GPT_M), blk128, 0, stream>>>(x, ln1_w + l * GPT_C,
                                                  ln1_b + l * GPT_C, hbuf);
    // qkv = h @ attn_w^T  (bf16 out)
    gemm_wmma<GOP_BF16><<<dim3(3 * GPT_C / 64, mblocks), blk256, 0, stream>>>(
        hbuf, GPT_C, aw, GPT_C, GPT_M, 3 * GPT_C, GPT_C, qkv, 3 * GPT_C,
        nullptr);
    // y = attention(qkv)
    attn_kernel<<<dim3(GPT_B * GPT_H * (GPT_T / 16)), blk32, 0, stream>>>(qkv,
                                                                          ybuf);
    // x = x + y @ proj_w^T
    gemm_wmma<GOP_RESID><<<dim3(GPT_C / 64, mblocks), blk256, 0, stream>>>(
        ybuf, GPT_C, pw, GPT_C, GPT_M, GPT_C, GPT_C, x, GPT_C, x);
    // h = LN2(x)
    ln_kernel<<<dim3(GPT_M), blk128, 0, stream>>>(x, ln2_w + l * GPT_C,
                                                  ln2_b + l * GPT_C, hbuf);
    // a1 = gelu(h @ fc1^T)  (bf16 out)
    gemm_wmma<GOP_GELU><<<dim3(4 * GPT_C / 64, mblocks), blk256, 0, stream>>>(
        hbuf, GPT_C, f1, GPT_C, GPT_M, 4 * GPT_C, GPT_C, a1, 4 * GPT_C,
        nullptr);
    // x = x + a1 @ fc2^T
    gemm_wmma<GOP_RESID><<<dim3(GPT_C / 64, mblocks), blk256, 0, stream>>>(
        a1, 4 * GPT_C, f2, 4 * GPT_C, GPT_M, GPT_C, 4 * GPT_C, x, GPT_C, x);
  }

  // final LN + tied LM head
  ln_kernel<<<dim3(GPT_M), blk128, 0, stream>>>(x, lnf_w, lnf_b, hbuf);
  gemm_wmma<GOP_F32><<<dim3((GPT_V + 63) / 64, mblocks), blk256, 0, stream>>>(
      hbuf, GPT_C, tok_emb, GPT_C, GPT_M, GPT_V, GPT_C, d_out, GPT_V, nullptr);
}
